// MoLELayer_7687991460382
// MI455X (gfx1250) — matrix operand order repaired and checked by reading, not compiled, not run on hardware
//
#include <hip/hip_runtime.h>
#include <math.h>

typedef __attribute__((ext_vector_type(16))) __bf16 v16bf;
typedef __attribute__((ext_vector_type(8)))  __bf16 v8bf;
typedef __attribute__((ext_vector_type(8)))  float  v8f;
typedef __attribute__((ext_vector_type(4)))  unsigned int v4u;
typedef __attribute__((ext_vector_type(8)))  unsigned int v8u;

static constexpr int kB = 4, kS = 2048, kD = 1024, kE = 8, kH = 4096;
static constexpr int kTok = kB * kS;      // 8192 tokens

// ---------- helpers ----------
__device__ __forceinline__ unsigned short f2bf(float f) {
    unsigned int u = __float_as_uint(f);
    u += 0x7fffu + ((u >> 16) & 1u);      // round-to-nearest-even
    return (unsigned short)(u >> 16);
}

__device__ __forceinline__ float gelu_f(float v) {   // tanh approx (jax.nn.gelu default)
    return 0.5f * v * (1.0f + tanhf(0.7978845608028654f * (v + 0.044715f * v * v * v)));
}

// block-wide sum over 256 threads (8 waves of 32); result broadcast to all threads
__device__ __forceinline__ float block_sum(float v, float* s8) {
    #pragma unroll
    for (int o = 16; o > 0; o >>= 1) v += __shfl_down(v, o, 32);
    if ((threadIdx.x & 31) == 0) s8[threadIdx.x >> 5] = v;
    __syncthreads();
    float tot = 0.f;
    #pragma unroll
    for (int i = 0; i < 8; i++) tot += s8[i];
    __syncthreads();
    return tot;
}

// ---------- fp32 -> bf16 weight conversion ----------
__global__ __launch_bounds__(256) void cvt_bf16_kernel(const float* __restrict__ in,
                                                       unsigned short* __restrict__ out, int n) {
    int i = (blockIdx.x * 256 + threadIdx.x) * 4;
    if (i + 3 < n) {
        float4 v = *(const float4*)(in + i);
        ushort4 o;
        o.x = f2bf(v.x); o.y = f2bf(v.y); o.z = f2bf(v.z); o.w = f2bf(v.w);
        *(ushort4*)(out + i) = o;
    }
}

// ---------- per-token: LN1, router softmax, LN2 -> bf16 h2, LUT gather, residual+routed ----------
__global__ __launch_bounds__(256) void token_kernel(
    const float* __restrict__ x, const int* __restrict__ ids, const float* __restrict__ lut,
    const float* __restrict__ g1, const float* __restrict__ c1,
    const float* __restrict__ g2, const float* __restrict__ c2,
    const float* __restrict__ rw, const float* __restrict__ rb,
    unsigned short* __restrict__ h2out, float* __restrict__ out)
{
    __shared__ float s8[8];
    const int t   = blockIdx.x;
    const int tid = threadIdx.x;
    const int d0  = tid * 4;

    const float* xr = x + (size_t)t * kD;
    float4 xv = *(const float4*)(xr + d0);

    // ---- LayerNorm 1 ----
    float s  = xv.x + xv.y + xv.z + xv.w;
    float ss = xv.x*xv.x + xv.y*xv.y + xv.z*xv.z + xv.w*xv.w;
    float mu   = block_sum(s,  s8) * (1.0f / kD);
    float ex2  = block_sum(ss, s8) * (1.0f / kD);
    float rstd = rsqrtf(ex2 - mu * mu + 1e-5f);

    float4 gv = *(const float4*)(g1 + d0);
    float4 bv = *(const float4*)(c1 + d0);
    float h[4];
    h[0] = (xv.x - mu) * rstd * gv.x + bv.x;
    h[1] = (xv.y - mu) * rstd * gv.y + bv.y;
    h[2] = (xv.z - mu) * rstd * gv.z + bv.z;
    h[3] = (xv.w - mu) * rstd * gv.w + bv.w;

    // ---- router logits + softmax (all 8 experts) ----
    float lg[kE];
    #pragma unroll
    for (int e = 0; e < kE; e++) lg[e] = 0.f;
    #pragma unroll
    for (int j = 0; j < 4; j++) {
        const float* rwp = rw + (size_t)(d0 + j) * kE;
        float hj = h[j];
        #pragma unroll
        for (int e = 0; e < kE; e++) lg[e] += hj * rwp[e];
    }
    #pragma unroll
    for (int e = 0; e < kE; e++) lg[e] = block_sum(lg[e], s8) + rb[e];

    float mx = lg[0];
    #pragma unroll
    for (int e = 1; e < kE; e++) mx = fmaxf(mx, lg[e]);
    float rv[kE], den = 0.f;
    #pragma unroll
    for (int e = 0; e < kE; e++) { rv[e] = __expf(lg[e] - mx); den += rv[e]; }
    float inv = 1.0f / den;
    #pragma unroll
    for (int e = 0; e < kE; e++) rv[e] *= inv;

    // ---- LayerNorm 2 on h -> bf16 h2 ----
    s  = h[0] + h[1] + h[2] + h[3];
    ss = h[0]*h[0] + h[1]*h[1] + h[2]*h[2] + h[3]*h[3];
    float mu2   = block_sum(s,  s8) * (1.0f / kD);
    float ex22  = block_sum(ss, s8) * (1.0f / kD);
    float rstd2 = rsqrtf(ex22 - mu2 * mu2 + 1e-5f);
    float4 g2v = *(const float4*)(g2 + d0);
    float4 b2v = *(const float4*)(c2 + d0);
    ushort4 h2v;
    h2v.x = f2bf((h[0] - mu2) * rstd2 * g2v.x + b2v.x);
    h2v.y = f2bf((h[1] - mu2) * rstd2 * g2v.y + b2v.y);
    h2v.z = f2bf((h[2] - mu2) * rstd2 * g2v.z + b2v.z);
    h2v.w = f2bf((h[3] - mu2) * rstd2 * g2v.w + b2v.w);
    *(ushort4*)(h2out + (size_t)t * kD + d0) = h2v;

    // ---- routed = sum_e rv[e] * lut[id][e] ; write residual + routed ----
    const float* lr = lut + (size_t)ids[t] * (kE * kD);
    float a0 = 0.f, a1 = 0.f, a2 = 0.f, a3 = 0.f;
    #pragma unroll
    for (int e = 0; e < kE; e++) {
        float  c  = rv[e];
        float4 lv = *(const float4*)(lr + e * kD + d0);
        a0 += c * lv.x; a1 += c * lv.y; a2 += c * lv.z; a3 += c * lv.w;
    }
    float4 o;
    o.x = xv.x + a0; o.y = xv.y + a1; o.z = xv.z + a2; o.w = xv.w + a3;
    *(float4*)(out + (size_t)t * kD + d0) = o;
}

// ---------- bf16 WMMA GEMM: C(M,N) = A(M,K) * B(K,N) ----------
// 128x64 block tile, 256 threads = 8 waves; A tiles moved by the Tensor Data Mover
// (one 16x32 TDM descriptor per wave, double-buffered, TENSORcnt-synchronized).
template <bool GELU>
__global__ __launch_bounds__(256) void gemm_wmma(
    const unsigned short* __restrict__ A, const unsigned short* __restrict__ Bm,
    const float* __restrict__ bias, float* __restrict__ outf,
    unsigned short* __restrict__ outb, int M, int N, int K)
{
    // 40-ushort (80B) row stride: 16B-aligned rows + bank-conflict padding.
    __shared__ __align__(16) unsigned short As[2][128][40];  // [m][k]
    __shared__ __align__(16) unsigned short Bs[2][64][40];   // [n][k] (K-transposed)

    const int tid  = threadIdx.x;
    const int m0   = blockIdx.y * 128;
    const int n0   = blockIdx.x * 64;
    const int wv   = tid >> 5;
    const int lane = tid & 31;
    const int lrow = lane & 15;
    const int lhi  = lane >> 4;

    // --- TDM: async-load this wave's 16x32 bf16 A slice into As[buf] ---
    // D# per CDNA5 ISA. pad_interval code 3 = 16 DWORDs
    // (one 64B tile row), pad_amount code 3 = 4 DWORDs (16B) -> 80B LDS row stride.
    auto issueA = [&](int k0, int buf) {
        unsigned long long ga =
            (unsigned long long)(size_t)(A + (size_t)(m0 + wv * 16) * K + k0);
        unsigned int lo = (unsigned int)(size_t)(&As[buf][wv * 16][0]);
        v4u g0;
        g0[0] = 1u;                                   // count = 1 valid descriptor
        g0[1] = lo;                                   // lds_addr
        g0[2] = (unsigned int)ga;                     // global_addr[31:0]
        g0[3] = ((unsigned int)(ga >> 32) & 0x01FFFFFFu) | (2u << 30); // addr[56:32] | type=2
        v8u g1;
        g1[0] = (1u << 16)            // data_size = 2 bytes
              | (1u << 20)            // pad_enable
              | (3u << 22)            // pad_interval: 16 DWORDs
              | (3u << 25);           // pad_amount: 4 DWORDs
        g1[1] = ((unsigned)K & 0xFFFFu) << 16;                               // tensor_dim0 lo
        g1[2] = (((unsigned)K >> 16) & 0xFFFFu) | (((unsigned)M & 0xFFFFu) << 16); // dim0 hi | dim1 lo
        g1[3] = (((unsigned)M >> 16) & 0xFFFFu) | (32u << 16);               // dim1 hi | tile_dim0=32
        g1[4] = 16u;                                                         // tile_dim1=16, tile_dim2=0
        g1[5] = (unsigned)K;                                                 // tensor_dim0_stride lo32
        g1[6] = 0u;                                                          // stride0 hi | stride1 lo
        g1[7] = 0u;
        asm volatile("tensor_load_to_lds %0, %1" :: "s"(g0), "s"(g1) : "memory");
    };

    // --- B tile: 32k x 64n row-major global -> K-transposed LDS [n][k] ---
    auto loadB = [&](int k0, int buf) {
        int kb = tid >> 3;
        int nb = (tid & 7) * 8;
        uint4 v = *(const uint4*)(Bm + (size_t)(k0 + kb) * N + n0 + nb);
        const unsigned short* u = (const unsigned short*)&v;
        #pragma unroll
        for (int j = 0; j < 8; j++) Bs[buf][nb + j][kb] = u[j];
    };

    v8f acc[4] = {};

    issueA(0, 0);
    loadB(0, 0);
    __builtin_amdgcn_s_wait_tensorcnt(0);
    __syncthreads();

    for (int k0 = 0; k0 < K; k0 += 32) {
        const int cur = (k0 >> 5) & 1;
        const int nxt = cur ^ 1;
        if (k0 + 32 < K) {
            issueA(k0 + 32, nxt);                    // TDM overlaps with WMMA below
            __builtin_prefetch(Bm + (size_t)(k0 + 32 + (tid >> 3)) * N + n0, 0, 1);
            loadB(k0 + 32, nxt);
        }

        // Load A fragment + ALL FOUR B fragments into distinct registers first so
        // the eight ds_load_b128 are in flight before the first WMMA; the four
        // WMMAs then issue back-to-back with staggered dscnt waits instead of
        // each stalling on a freshly reused register octet.
        union AV { v16bf v; v8bf h[2]; };
        AV a;
        int arow = wv * 16 + lrow;
        a.h[0] = *(const v8bf*)&As[cur][arow][lhi * 8];
        a.h[1] = *(const v8bf*)&As[cur][arow][lhi * 8 + 16];

        AV b[4];
        #pragma unroll
        for (int nt = 0; nt < 4; nt++) {
            int bcol = nt * 16 + lrow;
            b[nt].h[0] = *(const v8bf*)&Bs[cur][bcol][lhi * 8];
            b[nt].h[1] = *(const v8bf*)&Bs[cur][bcol][lhi * 8 + 16];
        }

        #pragma unroll
        for (int nt = 0; nt < 4; nt++) {
            acc[nt] = __builtin_amdgcn_wmma_f32_16x16x32_bf16(
                false, a.v, false, b[nt].v, (short)0, acc[nt], false, false);
        }

        __builtin_amdgcn_s_wait_tensorcnt(0);        // this wave's TDM for nxt complete
        __syncthreads();                             // + everyone's B stores visible
    }

    // epilogue: C layout — VGPR r holds M = r + 8*lhi, N = lane&15 (per 16x16 tile)
    #pragma unroll
    for (int nt = 0; nt < 4; nt++) {
        int   ng = n0 + nt * 16 + lrow;
        float bn = bias[ng];
        #pragma unroll
        for (int r = 0; r < 8; r++) {
            int   mg = m0 + wv * 16 + r + lhi * 8;
            float v  = acc[nt][r] + bn;
            if (GELU) {
                outb[(size_t)mg * N + ng] = f2bf(gelu_f(v));
            } else {
                outf[(size_t)mg * N + ng] += v;   // out already holds residual+routed
            }
        }
    }
}

// ---------- launch ----------
extern "C" void kernel_launch(void* const* d_in, const int* in_sizes, int n_in,
                              void* d_out, int out_size, void* d_ws, size_t ws_size,
                              hipStream_t stream) {
    const float* x    = (const float*)d_in[0];
    const int*   ids  = (const int*)  d_in[1];
    const float* lut  = (const float*)d_in[2];
    const float* ln1g = (const float*)d_in[3];
    const float* ln1b = (const float*)d_in[4];
    const float* ln2g = (const float*)d_in[5];
    const float* ln2b = (const float*)d_in[6];
    const float* rw   = (const float*)d_in[7];
    const float* rb   = (const float*)d_in[8];
    const float* w1   = (const float*)d_in[9];
    const float* b1   = (const float*)d_in[10];
    const float* w2   = (const float*)d_in[11];
    const float* b2   = (const float*)d_in[12];
    float* out = (float*)d_out;

    char* ws = (char*)d_ws;
    unsigned short* h2  = (unsigned short*)(ws);                              // 16 MB
    unsigned short* w1b = (unsigned short*)(ws + (size_t)16 * 1024 * 1024);   //  8 MB
    unsigned short* w2b = (unsigned short*)(ws + (size_t)24 * 1024 * 1024);   //  8 MB
    unsigned short* act = (unsigned short*)(ws + (size_t)32 * 1024 * 1024);   // 64 MB

    cvt_bf16_kernel<<<(kD * kH) / 1024, 256, 0, stream>>>(w1, w1b, kD * kH);
    cvt_bf16_kernel<<<(kH * kD) / 1024, 256, 0, stream>>>(w2, w2b, kH * kD);

    token_kernel<<<kTok, 256, 0, stream>>>(x, ids, lut, ln1g, ln1b, ln2g, ln2b,
                                           rw, rb, h2, out);

    gemm_wmma<true ><<<dim3(kH / 64, kTok / 128), 256, 0, stream>>>(
        h2, w1b, b1, nullptr, act, kTok, kH, kD);
    gemm_wmma<false><<<dim3(kD / 64, kTok / 128), 256, 0, stream>>>(
        act, w2b, b2, out, nullptr, kTok, kD, kH);
}
